// AttentionBlock_63136019251204
// MI455X (gfx1250) — compile-verified, matrix operand order checked by reading, not run on hardware
//
#include <hip/hip_runtime.h>

// ---------------------------------------------------------------------------
// Causal attention block for MI455X (gfx1250), wave32 + WMMA f16 16x16x32.
//   B=4, T=4096, D=256, K=V=64.  out = concat(x, softmax(QK^T/8, causal) V)
//
// Attention is computed TRANSPOSED (S^T = K Q^T, acc^T = V^T P^T) so that:
//   - each lane owns one query column -> softmax state is per-lane scalar,
//     no alpha/l broadcasts, row reductions = 1 shfl_xor(16) each;
//   - P^T is an identity register remap of the S^T WMMA accumulators
//     (transpose-symmetric fragment layouts) -> zero LDS traffic;
//   - final stores are contiguous 16B per lane.
// Softmax is done in base 2 (v_exp_f32 is exp2): log2(e)/sqrt(64) is folded
// into the Q projection, which leaves probabilities bit-identical in exact
// arithmetic.
// ---------------------------------------------------------------------------

typedef __attribute__((ext_vector_type(16))) _Float16 v16h;
typedef __attribute__((ext_vector_type(8)))  _Float16 v8h;
typedef __attribute__((ext_vector_type(8)))  float    v8f;

#define NROWS   16384   // B*T
#define DIMS    256
#define KV      64
#define TT      4096
#define OUTW    320     // DIMS + V
// 1/sqrt(64) * log2(e): base-2 softmax scale folded into Q
#define QSCALE  0.1803368801111204f

__device__ __forceinline__ v8f wmma_f16(v16h a, v16h b, v8f c) {
    return __builtin_amdgcn_wmma_f32_16x16x32_f16(false, a, false, b,
                                                  (short)0, c, false, false);
}

// ---------------------------------------------------------------------------
// Kernel 1: QKV projection.  grid = (NROWS/16, 3), block = 128 (4 waves).
// Wave w computes a 16x16 tile of X @ W (N-tile w of 4), K=256 in 8 steps.
// Q is pre-scaled by QSCALE.  V is stored TRANSPOSED per batch
// (Vt[b][vcol][key]) so the attention A-fragments load as contiguous 16B runs.
// ---------------------------------------------------------------------------
__global__ __launch_bounds__(128)
void qkv_proj_kernel(const float* __restrict__ x,
                     const float* __restrict__ qw,
                     const float* __restrict__ kw,
                     const float* __restrict__ vw,
                     _Float16* __restrict__ qf,
                     _Float16* __restrict__ kf,
                     _Float16* __restrict__ vt) {
    const int lane = threadIdx.x & 31;
    const int wave = threadIdx.x >> 5;
    const int m0   = blockIdx.x << 4;       // row-tile base in [0, NROWS)
    const int n0   = wave << 4;             // col-tile base in [0, 64)
    const int sel  = blockIdx.y;            // 0=Q, 1=K, 2=V
    const float* __restrict__ wmat = (sel == 0) ? qw : (sel == 1) ? kw : vw;
    const float scale = (sel == 0) ? QSCALE : 1.0f;

    const int mr = lane & 15;               // A-matrix row / B,C column
    const int h  = lane >> 4;

    v8f c = {};
    for (int kc = 0; kc < DIMS; kc += 32) {
        v16h a, b;
#pragma unroll
        for (int g = 0; g < 2; ++g) {
            // A: 8 contiguous k's = two float4 loads
            const float4 xa = *(const float4*)(x + (size_t)(m0 + mr) * DIMS + kc + (g << 4) + (h << 3));
            const float4 xb = *(const float4*)(x + (size_t)(m0 + mr) * DIMS + kc + (g << 4) + (h << 3) + 4);
            a[(g << 3) + 0] = (_Float16)xa.x; a[(g << 3) + 1] = (_Float16)xa.y;
            a[(g << 3) + 2] = (_Float16)xa.z; a[(g << 3) + 3] = (_Float16)xa.w;
            a[(g << 3) + 4] = (_Float16)xb.x; a[(g << 3) + 5] = (_Float16)xb.y;
            a[(g << 3) + 6] = (_Float16)xb.z; a[(g << 3) + 7] = (_Float16)xb.w;
            // B: W[k][n], stride KV between consecutive k
#pragma unroll
            for (int i = 0; i < 8; ++i) {
                const int k = kc + (g << 4) + (h << 3) + i;
                b[(g << 3) + i] = (_Float16)wmat[(size_t)k * KV + n0 + mr];
            }
        }
        c = wmma_f16(a, b, c);
    }
    if (sel == 2) {
        // Vt[b][vcol][key]: C-layout lane holds vcol = n0+mr, rows v+8h
#pragma unroll
        for (int v = 0; v < 8; ++v) {
            const int row = m0 + v + (h << 3);       // global row in [0, B*T)
            const int bb  = row >> 12;               // / TT
            const int key = row & (TT - 1);
            vt[((size_t)bb * KV + n0 + mr) * TT + key] = (_Float16)c[v];
        }
    } else {
        _Float16* __restrict__ omat = (sel == 0) ? qf : kf;
#pragma unroll
        for (int v = 0; v < 8; ++v) {
            const int row = m0 + v + (h << 3);
            omat[(size_t)row * KV + n0 + mr] = (_Float16)(c[v] * scale);
        }
    }
}

// ---------------------------------------------------------------------------
// Kernel 2: copy inputs into out[:, 0:256] (float4 streams).
// ---------------------------------------------------------------------------
__global__ __launch_bounds__(256)
void copy_input_kernel(const float* __restrict__ x, float* __restrict__ out) {
    const int i = blockIdx.x * 256 + threadIdx.x;     // one float4 each
    const int row = i >> 6;                            // 64 float4 per row
    const int c   = i & 63;
    ((float4*)out)[(size_t)row * (OUTW / 4) + c] =
        ((const float4*)x)[(size_t)row * (DIMS / 4) + c];
}

// ---------------------------------------------------------------------------
// Kernel 3: transposed flash attention.  grid = (T/64, B), block = 128.
// Each wave owns 16 query columns (q0..q0+15), loops over 64-key blocks.
// Per block: 8 WMMAs  S^T[t] = K_t  Q^T   (t = four 16-key tiles)
//            8 WMMAs  acc^T[t] += V^T_t P^T
// Softmax per query column: in-register + one shfl_xor(16) per reduction.
// No LDS, no workgroup barriers (waves diverge causally).
// ---------------------------------------------------------------------------
__global__ __launch_bounds__(128)
void attn_kernel(const _Float16* __restrict__ qf,
                 const _Float16* __restrict__ kf,
                 const _Float16* __restrict__ vtf,
                 float* __restrict__ out) {
    const int lane = threadIdx.x & 31;
    const int w    = threadIdx.x >> 5;
    const int b    = blockIdx.y;
    const int q0   = blockIdx.x * 64 + (w << 4);     // query base within batch
    const int n    = lane & 15;                      // this lane's query col
    const int h    = lane >> 4;
    const float NEG_INF = -__builtin_inff();

    const _Float16* __restrict__ Q  = qf  + (size_t)b * TT * KV;
    const _Float16* __restrict__ K  = kf  + (size_t)b * TT * KV;
    const _Float16* __restrict__ Vt = vtf + (size_t)b * KV * TT;  // [vcol][key]

    // Q^T B-fragments (loop-invariant): lane holds query column q0+n,
    // contraction dim d in two 32-chunks, 16B loads.
    v16h bq[2];
#pragma unroll
    for (int c = 0; c < 2; ++c)
#pragma unroll
        for (int g = 0; g < 2; ++g) {
            const v8h qv = *(const v8h*)(Q + (size_t)(q0 + n) * KV + (c << 5) + (g << 4) + (h << 3));
#pragma unroll
            for (int i = 0; i < 8; ++i) bq[c][(g << 3) + i] = qv[i];
        }

    // Per-lane softmax state for query q0+n (replicated across wave halves).
    float mA = NEG_INF, lA = 0.0f;
    v8f acc[4] = {v8f{}, v8f{}, v8f{}, v8f{}};       // acc^T: vcol tiles
    const int nkb = ((q0 + 15) >> 6) + 1;            // causal 64-key blocks

    for (int j = 0; j < nkb; ++j) {
        const int kb = j << 6;

        // Prefetch next key block (global_prefetch_b8)
        if (j + 1 < nkb) {
            const int kb2 = kb + 64;
            __builtin_prefetch(K + (size_t)(kb2 + lane) * KV, 0, 3);
            __builtin_prefetch(K + (size_t)(kb2 + 32 + lane) * KV, 0, 3);
            __builtin_prefetch(Vt + (size_t)(lane << 1) * TT + kb2, 0, 3);
            __builtin_prefetch(Vt + (size_t)((lane << 1) + 1) * TT + kb2, 0, 3);
        }

        // ---- S^T tiles: st[t] = K_t (16 keys x 64d) x Q^T (64d x 16q)
        v8f st[4];
#pragma unroll
        for (int t = 0; t < 4; ++t) {
            v8f sc = {};
#pragma unroll
            for (int c = 0; c < 2; ++c) {
                v16h ak;                              // A-frag: K tile rows
#pragma unroll
                for (int g = 0; g < 2; ++g) {
                    const v8h kv2 = *(const v8h*)(K + (size_t)(kb + (t << 4) + n) * KV + (c << 5) + (g << 4) + (h << 3));
#pragma unroll
                    for (int i = 0; i < 8; ++i) ak[(g << 3) + i] = kv2[i];
                }
                sc = wmma_f16(ak, bq[c], sc);
            }
            st[t] = sc;
        }

        // ---- causal mask (key = kb+t*16+v+8h vs query q0+n), diagonal only
        if (kb + 63 > q0) {
            const int q = q0 + n;
#pragma unroll
            for (int t = 0; t < 4; ++t)
#pragma unroll
                for (int v = 0; v < 8; ++v)
                    if (kb + (t << 4) + v + (h << 3) > q) st[t][v] = NEG_INF;
        }

        // ---- column max: 31 in-register + 1 cross-half shuffle
        float mx = st[0][0];
#pragma unroll
        for (int t = 0; t < 4; ++t)
#pragma unroll
            for (int v = 0; v < 8; ++v) mx = fmaxf(mx, st[t][v]);
        mx = fmaxf(mx, __shfl_xor(mx, 16));
        const float mnew  = fmaxf(mA, mx);
        const float alpha = (mA == NEG_INF) ? 0.0f : __builtin_amdgcn_exp2f(mA - mnew);
        mA = mnew;

        // ---- exponentiate (base 2, scale pre-folded) + column sum
        float rs = 0.0f;
#pragma unroll
        for (int t = 0; t < 4; ++t)
#pragma unroll
            for (int v = 0; v < 8; ++v) {
                st[t][v] = __builtin_amdgcn_exp2f(st[t][v] - mnew);  // exp2(-inf)=0
                rs += st[t][v];
            }
        rs += __shfl_xor(rs, 16);
        lA = lA * alpha + rs;

        // ---- P^T B-fragments: identity remap of S^T C-tiles
        //      (B chunk c over keys = concat of C-tiles 2c, 2c+1 per lane)
        v16h pf[2];
#pragma unroll
        for (int c = 0; c < 2; ++c)
#pragma unroll
            for (int i = 0; i < 8; ++i) {
                pf[c][i]     = (_Float16)st[2 * c][i];
                pf[c][8 + i] = (_Float16)st[2 * c + 1][i];
            }

        // ---- rescale accumulators by per-lane alpha (no broadcast needed)
#pragma unroll
        for (int t = 0; t < 4; ++t)
#pragma unroll
            for (int v = 0; v < 8; ++v) acc[t][v] *= alpha;

        // ---- acc^T[t] += V^T_t (16 vcols x 64 keys) x P^T (64 keys x 16q)
#pragma unroll
        for (int t = 0; t < 4; ++t) {
#pragma unroll
            for (int c = 0; c < 2; ++c) {
                v16h av;                              // A-frag: V^T tile rows
#pragma unroll
                for (int g = 0; g < 2; ++g) {
                    const v8h vv = *(const v8h*)(Vt + (size_t)((t << 4) + n) * TT + kb + (c << 5) + (g << 4) + (h << 3));
#pragma unroll
                    for (int i = 0; i < 8; ++i) av[(g << 3) + i] = vv[i];
                }
                acc[t] = wmma_f16(av, pf[c], acc[t]);
            }
        }
    }

    // ---- normalize (per-lane l) and store: one output row per lane,
    //      columns 256 + t*16 + 8h + v are contiguous -> 16B stores
    const float rl = 1.0f / lA;
    float* __restrict__ orow = out + ((size_t)b * TT + q0 + n) * OUTW + DIMS + (h << 3);
#pragma unroll
    for (int t = 0; t < 4; ++t) {
        float4 f0, f1;
        f0.x = acc[t][0] * rl; f0.y = acc[t][1] * rl;
        f0.z = acc[t][2] * rl; f0.w = acc[t][3] * rl;
        f1.x = acc[t][4] * rl; f1.y = acc[t][5] * rl;
        f1.z = acc[t][6] * rl; f1.w = acc[t][7] * rl;
        *(float4*)(orow + (t << 4))     = f0;
        *(float4*)(orow + (t << 4) + 4) = f1;
    }
}

// ---------------------------------------------------------------------------
// Host-side launch.  Input dict order: inputs, key_w, query_w, value_w.
// Workspace: Q/K f16 [B*T,64] + Vt f16 [B,64,T] = 6 MB.
// ---------------------------------------------------------------------------
extern "C" void kernel_launch(void* const* d_in, const int* in_sizes, int n_in,
                              void* d_out, int out_size, void* d_ws, size_t ws_size,
                              hipStream_t stream) {
    (void)in_sizes; (void)n_in; (void)out_size; (void)ws_size;

    const float* x  = (const float*)d_in[0];
    const float* kw = (const float*)d_in[1];
    const float* qw = (const float*)d_in[2];
    const float* vw = (const float*)d_in[3];
    float* out = (float*)d_out;

    _Float16* qf = (_Float16*)d_ws;
    _Float16* kf = qf + (size_t)NROWS * KV;
    _Float16* vt = kf + (size_t)NROWS * KV;

    // 1) QKV projections (WMMA); Q pre-scaled by log2(e)/sqrt(64); V transposed
    qkv_proj_kernel<<<dim3(NROWS / 16, 3), 128, 0, stream>>>(x, qw, kw, vw, qf, kf, vt);

    // 2) out[:, 0:256] = inputs
    copy_input_kernel<<<(NROWS * (DIMS / 4)) / 256, 256, 0, stream>>>(x, out);

    // 3) transposed flash attention -> out[:, 256:320]
    attn_kernel<<<dim3(TT / 64, 4), 128, 0, stream>>>(qf, kf, vt, out);
}
